// GraphAttentionLayer_9371618640198
// MI455X (gfx1250) — matrix operand order repaired and checked by reading, not compiled, not run on hardware
//
#include <hip/hip_runtime.h>
#include <hip/hip_bf16.h>

typedef __attribute__((ext_vector_type(16))) __bf16 v16bf;
typedef __attribute__((ext_vector_type(8)))  float  v8f;

#define NROW 8192
#define INF_ 256
#define OUTF 128
#define ALPHA 0.2f

// ---------- WMMA fragment helpers (CDNA5 wave32; A/B use contiguous k = 16*half + e, C/D per ISA 7.12.2) ----------

// A-matrix 16x32 bf16 from row-major f32 tile (ld = row stride in elems).
// lane<16: row=lane, elems 0..15 = K(0..15)
// lane>=16: row=lane-16, elems 0..15 = K(16..31)
__device__ inline v16bf load_A_bf16_from_f32(const float* tile, int ld, int lane) {
  int half = lane >> 4;
  int r = lane & 15;
  const float* p = tile + (long)r * ld + half * 16;
  v16bf a;
#pragma unroll
  for (int j = 0; j < 8; ++j) a[j] = (__bf16)p[j];
#pragma unroll
  for (int j = 0; j < 8; ++j) a[8 + j] = (__bf16)p[8 + j];
  return a;
}

// B-matrix 32x16 bf16 from row-major f32 tile (K x N, ld = row stride).
// lane<16: col=lane, elems 0..15 = K(0..15); lane>=16: col=lane-16, K(16..31)
__device__ inline v16bf load_B_bf16_from_f32(const float* tile, int ld, int lane) {
  int half = lane >> 4;
  int c = lane & 15;
  const float* p = tile + (long)(half * 16) * ld + c;
  v16bf b;
#pragma unroll
  for (int j = 0; j < 16; ++j) b[j] = (__bf16)p[(long)j * ld];
  return b;
}

// C/D 16x16 f32: lane<16: col=lane, vgpr j = row j; lane>=16: col=lane-16, row 8+j
__device__ inline void store_C_f32(float* tile, int ld, int lane, v8f c) {
  int half = lane >> 4;
  int col = lane & 15;
  float* p = tile + (long)(half * 8) * ld + col;
#pragma unroll
  for (int j = 0; j < 8; ++j) p[(long)j * ld] = c[j];
}

// ---------- Kernel 1: h = x @ W  (8192x256 @ 256x128), bf16 WMMA ----------
// 512 blocks x 128 threads (4 waves); block owns 16 rows, wave owns 2 N-tiles.
__global__ void gemm_h_kernel(const float* __restrict__ x,
                              const float* __restrict__ W,
                              float* __restrict__ h) {
  int lane = threadIdx.x & 31;
  int wave = threadIdx.x >> 5;
  int row0 = blockIdx.x * 16;
  int n0 = wave * 32;
  v8f acc0 = {};
  v8f acc1 = {};
  for (int k0 = 0; k0 < INF_; k0 += 32) {
    v16bf a  = load_A_bf16_from_f32(x + (long)row0 * INF_ + k0, INF_, lane);
    v16bf b0 = load_B_bf16_from_f32(W + (long)k0 * OUTF + n0, OUTF, lane);
    v16bf b1 = load_B_bf16_from_f32(W + (long)k0 * OUTF + n0 + 16, OUTF, lane);
    acc0 = __builtin_amdgcn_wmma_f32_16x16x32_bf16(false, a, false, b0, (short)0, acc0, false, false);
    acc1 = __builtin_amdgcn_wmma_f32_16x16x32_bf16(false, a, false, b1, (short)0, acc1, false, false);
  }
  store_C_f32(h + (long)row0 * OUTF + n0, OUTF, lane, acc0);
  store_C_f32(h + (long)row0 * OUTF + n0 + 16, OUTF, lane, acc1);
}

// ---------- Kernel 2: f_src[i] = h[i]·a[0:128], f_dst[i] = h[i]·a[128:256] ----------
__global__ void fvec_kernel(const float* __restrict__ h, const float* __restrict__ a,
                            float* __restrict__ fs, float* __restrict__ fd) {
  int i = blockIdx.x * blockDim.x + threadIdx.x;
  if (i >= NROW) return;
  const float* hr = h + (long)i * OUTF;
  float s = 0.f, d = 0.f;
#pragma unroll 4
  for (int f = 0; f < OUTF; ++f) {
    float hv = hr[f];
    s += hv * a[f];
    d += hv * a[OUTF + f];
  }
  fs[i] = s;
  fd[i] = d;
}

// ---------- Kernel 2b: pack h into bf16 WMMA B-fragment records ----------
// hB layout: [kt (256)][nt (8)][lane (32)][16 bf16]  -> 32B contiguous per lane.
// Element e of record (kt,nt,lane): h[kt*32 + (lane>>4)*16 + e][nt*16 + (lane&15)]
__global__ void packh_kernel(const float* __restrict__ h, __bf16* __restrict__ hB) {
  int t = blockIdx.x * blockDim.x + threadIdx.x;   // 65536 = 2048 frags * 32 lanes
  int lane = t & 31;
  int frag = t >> 5;                               // kt*8 + nt
  int nt = frag & 7;
  int kt = frag >> 3;
  int half = lane >> 4;
  int c = lane & 15;
  const float* src = h + (long)(kt * 32 + half * 16) * OUTF + nt * 16 + c;
  v16bf b;
#pragma unroll
  for (int e = 0; e < 16; ++e) b[e] = (__bf16)src[(long)e * OUTF];
  *(v16bf*)(hB + ((long)frag * 32 + lane) * 16) = b;
}

// ---------- Kernel 3: Q[i][j] = exp(leakyrelu(fs[i]+fd[j])) * adj[i][j] ----------
__global__ void buildQ_kernel(const float* __restrict__ adj,
                              const float* __restrict__ fs,
                              const float* __restrict__ fd,
                              float* __restrict__ Q) {
  long idx = ((long)blockIdx.x * blockDim.x + threadIdx.x) * 4;
  int i = (int)(idx >> 13);
  int j = (int)(idx & (NROW - 1));
  float4 av = *(const float4*)(adj + idx);
  float4 dv = *(const float4*)(fd + j);
  float s0 = fs[i];
  float4 q;
  float t;
  t = s0 + dv.x; t = t > 0.f ? t : ALPHA * t; q.x = __expf(t) * av.x;
  t = s0 + dv.y; t = t > 0.f ? t : ALPHA * t; q.y = __expf(t) * av.y;
  t = s0 + dv.z; t = t > 0.f ? t : ALPHA * t; q.z = __expf(t) * av.z;
  t = s0 + dv.w; t = t > 0.f ? t : ALPHA * t; q.w = __expf(t) * av.w;
  *(float4*)(Q + idx) = q;
}

// ---------- Kernel 4: init u/v scaling state ----------
__global__ void init_kernel(float* __restrict__ v, float* __restrict__ vacc) {
  int j = blockIdx.x * blockDim.x + threadIdx.x;
  if (j < NROW) { v[j] = 1.0f; vacc[j] = 0.0f; }
}

// ---------- Kernel 5: u = 1 / (Q v)  (row sums) ----------
// 1024 blocks x 256 threads; each wave owns one row, float4 streaming + prefetch.
__global__ void rowpass_kernel(const float* __restrict__ Q,
                               const float* __restrict__ v,
                               float* __restrict__ u) {
  int lane = threadIdx.x & 31;
  int wave = threadIdx.x >> 5;
  int row = blockIdx.x * 8 + wave;
  const float* qr = Q + (long)row * NROW;
  float acc = 0.f;
  for (int t = 0; t < NROW / 128; ++t) {
    int j = t * 128 + lane * 4;
    int jp = j + 2048;
    if (jp < NROW) __builtin_prefetch(qr + jp, 0, 1);   // global_prefetch_b8
    float4 qv = *(const float4*)(qr + j);
    float4 vv = *(const float4*)(v + j);
    acc += qv.x * vv.x + qv.y * vv.y + qv.z * vv.z + qv.w * vv.w;
  }
#pragma unroll
  for (int off = 16; off > 0; off >>= 1) acc += __shfl_down(acc, off, 32);
  if (lane == 0) u[row] = 1.0f / acc;
}

// ---------- Kernel 6: vacc += Q^T u partials (column sums over an i-chunk) ----------
// grid (32 col-strips, 16 i-chunks) x 256 threads; coalesced column-strip reads.
__global__ void colpass_kernel(const float* __restrict__ Q,
                               const float* __restrict__ u,
                               float* __restrict__ vacc) {
  int j = blockIdx.x * 256 + threadIdx.x;
  int i0 = blockIdx.y * 512;
  const float* qp = Q + (long)i0 * NROW + j;
  float acc = 0.f;
#pragma unroll 4
  for (int i = 0; i < 512; ++i) {
    if (i + 8 < 512) __builtin_prefetch(qp + (long)(i + 8) * NROW, 0, 1);
    acc += qp[(long)i * NROW] * u[i0 + i];
  }
  atomicAdd(&vacc[j], acc);
}

// ---------- Kernel 7: v = 1/vacc, reset vacc ----------
__global__ void invert_kernel(float* __restrict__ v, float* __restrict__ vacc) {
  int j = blockIdx.x * blockDim.x + threadIdx.x;
  if (j < NROW) { v[j] = 1.0f / vacc[j]; vacc[j] = 0.0f; }
}

// ---------- Kernel 8: P = diag(u) Q diag(v) (in place over Q), h' = P @ h ----------
// 512 blocks x 128 threads; block owns 16 rows; wave owns 2 of 8 N-tiles of h'.
// In-place safety: every wave consumes its Q loads into registers before the
// barrier; wave 0 then overwrites the tile with scaled f32 P.
__global__ void final_kernel(float* QP,                       // aliased Q (read) / P (write)
                             const __bf16* __restrict__ hB,   // packed B fragments of h
                             const float* __restrict__ u,
                             const float* __restrict__ v,
                             float* __restrict__ hp) {
  int lane = threadIdx.x & 31;
  int wave = threadIdx.x >> 5;
  int row0 = blockIdx.x * 16;
  int half = lane >> 4;
  int r = lane & 15;
  float ur = u[row0 + r];
  int nt0 = wave * 2;
  const v16bf* hBf = (const v16bf*)hB;
  v8f acc0 = {};
  v8f acc1 = {};
  for (int k0 = 0; k0 < NROW; k0 += 32) {
    // ---- A fragment: P tile = ur * Q * v, 4x b128 loads per lane ----
    const float* qp = QP + (long)(row0 + r) * NROW + k0 + half * 16;
    float4 q0 = *(const float4*)(qp + 0);
    float4 q1 = *(const float4*)(qp + 4);
    float4 q2 = *(const float4*)(qp + 8);
    float4 q3 = *(const float4*)(qp + 12);
    float4 v0 = *(const float4*)(v + k0 + half * 16 + 0);
    float4 v1 = *(const float4*)(v + k0 + half * 16 + 4);
    float4 v2 = *(const float4*)(v + k0 + half * 16 + 8);
    float4 v3 = *(const float4*)(v + k0 + half * 16 + 12);
    float4 p0, p1, p2, p3;
    p0.x = ur * q0.x * v0.x; p0.y = ur * q0.y * v0.y; p0.z = ur * q0.z * v0.z; p0.w = ur * q0.w * v0.w;
    p1.x = ur * q1.x * v1.x; p1.y = ur * q1.y * v1.y; p1.z = ur * q1.z * v1.z; p1.w = ur * q1.w * v1.w;
    p2.x = ur * q2.x * v2.x; p2.y = ur * q2.y * v2.y; p2.z = ur * q2.z * v2.z; p2.w = ur * q2.w * v2.w;
    p3.x = ur * q3.x * v3.x; p3.y = ur * q3.y * v3.y; p3.z = ur * q3.z * v3.z; p3.w = ur * q3.w * v3.w;
    v16bf a;
    a[0]  = (__bf16)p0.x; a[1]  = (__bf16)p0.y; a[2]  = (__bf16)p0.z; a[3]  = (__bf16)p0.w;
    a[4]  = (__bf16)p1.x; a[5]  = (__bf16)p1.y; a[6]  = (__bf16)p1.z; a[7]  = (__bf16)p1.w;
    a[8]  = (__bf16)p2.x; a[9]  = (__bf16)p2.y; a[10] = (__bf16)p2.z; a[11] = (__bf16)p2.w;
    a[12] = (__bf16)p3.x; a[13] = (__bf16)p3.y; a[14] = (__bf16)p3.z; a[15] = (__bf16)p3.w;
    __syncthreads();               // all waves hold Q-derived values in regs now
    if (wave == 0) {               // overwrite Q tile with final f32 P (b128 stores)
      float* pp = QP + (long)(row0 + r) * NROW + k0 + half * 16;
      *(float4*)(pp + 0)  = p0;
      *(float4*)(pp + 4)  = p1;
      *(float4*)(pp + 8)  = p2;
      *(float4*)(pp + 12) = p3;
    }
    // ---- B fragments: packed bf16 records, one v16bf (2x b128) each ----
    long kt = (long)(k0 >> 5);
    v16bf b0 = hBf[(kt * 8 + nt0) * 32 + lane];
    v16bf b1 = hBf[(kt * 8 + nt0 + 1) * 32 + lane];
    acc0 = __builtin_amdgcn_wmma_f32_16x16x32_bf16(false, a, false, b0, (short)0, acc0, false, false);
    acc1 = __builtin_amdgcn_wmma_f32_16x16x32_bf16(false, a, false, b1, (short)0, acc1, false, false);
  }
  store_C_f32(hp + (long)row0 * OUTF + nt0 * 16, OUTF, lane, acc0);
  store_C_f32(hp + (long)row0 * OUTF + nt0 * 16 + 16, OUTF, lane, acc1);
}

// ---------------------------- launch ----------------------------

extern "C" void kernel_launch(void* const* d_in, const int* in_sizes, int n_in,
                              void* d_out, int out_size, void* d_ws, size_t ws_size,
                              hipStream_t stream) {
  const float* x   = (const float*)d_in[0];  // [8192,256]
  const float* adj = (const float*)d_in[1];  // [8192,8192]
  const float* W   = (const float*)d_in[2];  // [256,128]
  const float* a   = (const float*)d_in[3];  // [256,1]

  float* hp = (float*)d_out;                 // h_prime [8192,128], first output
  float* QP = hp + (long)NROW * OUTF;        // P region [8192,8192]; holds Q, then P

  char* ws = (char*)d_ws;
  float* h    = (float*)ws;                                   // 4 MB
  float* fs   = (float*)(ws + (size_t)NROW * OUTF * sizeof(float));
  float* fd   = fs + NROW;
  float* u    = fd + NROW;
  float* v    = u + NROW;
  float* vacc = v + NROW;
  __bf16* hB  = (__bf16*)(vacc + NROW);                       // 2 MB packed fragments
  // total ws use: 4 MB + 160 KB + 2 MB

  gemm_h_kernel<<<NROW / 16, 128, 0, stream>>>(x, W, h);
  fvec_kernel<<<NROW / 256, 256, 0, stream>>>(h, a, fs, fd);
  packh_kernel<<<256, 256, 0, stream>>>(h, hB);               // 65536 threads
  buildQ_kernel<<<(int)(((long)NROW * NROW) / (256 * 4)), 256, 0, stream>>>(adj, fs, fd, QP);
  init_kernel<<<NROW / 256, 256, 0, stream>>>(v, vacc);

  for (int it = 0; it < 5; ++it) {
    rowpass_kernel<<<NROW / 8, 256, 0, stream>>>(QP, v, u);                   // u = 1/(Q v)
    colpass_kernel<<<dim3(NROW / 256, 16), 256, 0, stream>>>(QP, u, vacc);    // vacc = Q^T u
    invert_kernel<<<NROW / 256, 256, 0, stream>>>(v, vacc);                   // v = 1/vacc
  }

  final_kernel<<<NROW / 16, 128, 0, stream>>>(QP, hB, u, v, hp);
}
